// HGRNBitMLPFixed_12386685681972
// MI455X (gfx1250) — 8-bit matrix operand order repaired and checked by reading, compiled, not run on hardware
//
#include <hip/hip_runtime.h>
#include <cstdint>
#include <cstddef>


typedef __attribute__((ext_vector_type(8))) int v8i;
typedef __attribute__((ext_vector_type(4))) int v4i;
typedef __attribute__((ext_vector_type(2))) int v2i;

#define M_TOK 8192
#define HID   2048
#define INTER 5632
#define NGATE 11264

union V8frag {
    v8i v;
    v2i d2[4];
    v4i d4[2];
};


__global__ __launch_bounds__(256)
void wabs_partial_kernel(const float* __restrict__ w, long n, float* __restrict__ part) {
    float s = 0.f;
    for (long i = (long)blockIdx.x * blockDim.x + threadIdx.x; i < n;
         i += (long)gridDim.x * blockDim.x)
        s += fabsf(w[i]);
    __shared__ float red[256];
    red[threadIdx.x] = s;
    __syncthreads();
    for (int st = 128; st > 0; st >>= 1) {
        if ((int)threadIdx.x < st) red[threadIdx.x] += red[threadIdx.x + st];
        __syncthreads();
    }
    if (threadIdx.x == 0) part[blockIdx.x] = red[0];
}

__global__ __launch_bounds__(256)
void wscale_finalize_kernel(const float* __restrict__ part, float* __restrict__ wsc) {
    __shared__ float red[256];
    const int b    = blockIdx.x;
    const int offp = b ? 2048 : 0;
    const int cnt  = b ? 1024 : 2048;
    float s = 0.f;
    for (int i = threadIdx.x; i < cnt; i += 256) s += part[offp + i];
    red[threadIdx.x] = s;
    __syncthreads();
    for (int st = 128; st > 0; st >>= 1) {
        if ((int)threadIdx.x < st) red[threadIdx.x] += red[threadIdx.x + st];
        __syncthreads();
    }
    if (threadIdx.x == 0) {
        double n   = b ? (double)HID * (double)INTER : (double)NGATE * (double)HID;
        float mean = (float)((double)red[0] / n);
        wsc[b] = fmaxf(mean, 1e-5f);
    }
}

__global__ __launch_bounds__(256)
void quant_w_kernel(const float* __restrict__ w, int8_t* __restrict__ wq, long n,
                    const float* __restrict__ wsc) {
    const float s = 1.f / wsc[0];
    for (long i = (long)blockIdx.x * blockDim.x + threadIdx.x; i < n;
         i += (long)gridDim.x * blockDim.x) {
        float q = fminf(fmaxf(rintf(w[i] * s), -1.f), 1.f);
        wq[i] = (int8_t)q;
    }
}


__global__ __launch_bounds__(256)
void act_quant_kernel(const float* __restrict__ x, int8_t* __restrict__ q,
                      float* __restrict__ deq, int W) {
    const int row = blockIdx.x;
    const float* xr = x + (size_t)row * W;
    float ss = 0.f, mx = 0.f;
    for (int i = threadIdx.x; i < W; i += 256) {
        float v = xr[i];
        ss += v * v;
        mx = fmaxf(mx, fabsf(v));
    }
    __shared__ float rs[256], rm[256];
    rs[threadIdx.x] = ss;
    rm[threadIdx.x] = mx;
    __syncthreads();
    for (int st = 128; st > 0; st >>= 1) {
        if ((int)threadIdx.x < st) {
            rs[threadIdx.x] += rs[threadIdx.x + st];
            rm[threadIdx.x] = fmaxf(rm[threadIdx.x], rm[threadIdx.x + st]);
        }
        __syncthreads();
    }
    const float rms  = rsqrtf(rs[0] / (float)W + 1e-6f);
    const float amax = rm[0] * rms;
    const float s    = 127.f / fmaxf(amax, 1e-5f);
    if (threadIdx.x == 0) deq[row] = 1.f / s;
    int8_t* qr = q + (size_t)row * W;
    for (int i = threadIdx.x; i < W; i += 256) {
        float v  = xr[i] * rms;
        float qv = fminf(fmaxf(rintf(v * s), -128.f), 127.f);
        qr[i] = (int8_t)qv;
    }
}


template <bool FUSED>
__global__ __launch_bounds__(256)
void bitgemm_kernel(const int8_t* __restrict__ Aq, const int8_t* __restrict__ Wq,
                    const float* __restrict__ a_deq, const float* __restrict__ w_deq,
                    float* __restrict__ out, int K) {
    constexpr int LDA   = 80;
    constexpr int LDB   = 80;
    constexpr int ABUF  = 128 * LDA;
    constexpr int BBUF  = 128 * LDB;
    __shared__ alignas(16) char ldsA[2][ABUF];
    __shared__ alignas(16) char ldsB[2][BBUF];

    const int t    = threadIdx.x;
    const int lane = t & 31;
    const int lrow = lane & 15;
    const int hi   = lane >> 4;
    const int wave = t >> 5;
    const int wm   = wave & 3;
    const int wn   = wave >> 2;

    const int m0    = blockIdx.y * 128;
    const int nbase = FUSED ? blockIdx.x * 64 : blockIdx.x * 128;

    const int chunk0 = t, chunk1 = t + 256;
    const int rA0 = chunk0 >> 2, cA0 = (chunk0 & 3) * 16;
    const int rA1 = chunk1 >> 2, cA1 = (chunk1 & 3) * 16;
    const int gn0 = FUSED ? (nbase + (rA0 & 63) + (rA0 >> 6) * INTER) : (nbase + rA0);
    const int gn1 = FUSED ? (nbase + (rA1 & 63) + (rA1 >> 6) * INTER) : (nbase + rA1);
    const int8_t* gA0 = Aq + (size_t)(m0 + rA0) * K + cA0;
    const int8_t* gA1 = Aq + (size_t)(m0 + rA1) * K + cA1;
    const int8_t* gB0 = Wq + (size_t)gn0 * K + cA0;
    const int8_t* gB1 = Wq + (size_t)gn1 * K + cA1;
    const unsigned lA0 = (unsigned)(uintptr_t)&ldsA[0][rA0 * LDA + cA0];
    const unsigned lA1 = (unsigned)(uintptr_t)&ldsA[0][rA1 * LDA + cA1];
    const unsigned lB0 = (unsigned)(uintptr_t)&ldsB[0][rA0 * LDB + cA0];
    const unsigned lB1 = (unsigned)(uintptr_t)&ldsB[0][rA1 * LDB + cA1];

    auto stage = [&](int buf, int kk) {
        const unsigned oA = buf ? (unsigned)ABUF : 0u;
        const unsigned oB = buf ? (unsigned)BBUF : 0u;
        asm volatile("global_load_async_to_lds_b128 %0, %1, off"
                     :: "v"(lA0 + oA), "v"(gA0 + kk) : "memory");
        asm volatile("global_load_async_to_lds_b128 %0, %1, off"
                     :: "v"(lA1 + oA), "v"(gA1 + kk) : "memory");
        asm volatile("global_load_async_to_lds_b128 %0, %1, off"
                     :: "v"(lB0 + oB), "v"(gB0 + kk) : "memory");
        asm volatile("global_load_async_to_lds_b128 %0, %1, off"
                     :: "v"(lB1 + oB), "v"(gB1 + kk) : "memory");
    };

    v8i acc[2][4];
    const v8i vz = {0, 0, 0, 0, 0, 0, 0, 0};
#pragma unroll
    for (int i = 0; i < 2; ++i)
#pragma unroll
        for (int j = 0; j < 4; ++j) acc[i][j] = vz;

    stage(0, 0);

    int buf = 0;
    for (int k0 = 0; k0 < K; k0 += 64, buf ^= 1) {
        asm volatile("s_wait_asynccnt 0x0" ::: "memory");
        __syncthreads();
        const int kn = (k0 + 64 < K) ? (k0 + 64) : 0;
        stage(buf ^ 1, kn);

        const char* bufA = ldsA[buf];
        const char* bufB = ldsB[buf];
        V8frag a[2], b[4];
#pragma unroll
        for (int mi = 0; mi < 2; ++mi) {
            const char* ap = bufA + (wm * 32 + mi * 16 + lrow) * LDA + hi * 16;
            a[mi].d4[0] = *(const v4i*)(ap);
            a[mi].d4[1] = *(const v4i*)(ap + 32);
        }
#pragma unroll
        for (int f = 0; f < 4; ++f) {
            int brow = FUSED ? ((f >> 1) * 64 + wn * 32 + (f & 1) * 16)
                             : (wn * 64 + f * 16);
            const char* bp = bufB + (brow + lrow) * LDB + hi * 16;
            b[f].d4[0] = *(const v4i*)(bp);
            b[f].d4[1] = *(const v4i*)(bp + 32);
        }

#pragma unroll
        for (int mi = 0; mi < 2; ++mi)
#pragma unroll
            for (int f = 0; f < 4; ++f)
                acc[mi][f] = __builtin_amdgcn_wmma_i32_16x16x64_iu8(
                    true, a[mi].v, true, b[f].v, acc[mi][f], false, false);
    }

    const float wsc = w_deq[0];
#pragma unroll
    for (int mi = 0; mi < 2; ++mi) {
        const int mb = m0 + wm * 32 + mi * 16 + hi * 8;
        if constexpr (FUSED) {
#pragma unroll
            for (int f = 0; f < 2; ++f) {
                const int ncol = nbase + wn * 32 + f * 16 + lrow;
#pragma unroll
                for (int i = 0; i < 8; ++i) {
                    const int m   = mb + i;
                    const float d = a_deq[m] * wsc;
                    const float g = (float)acc[mi][f][i] * d;
                    const float u = (float)acc[mi][f + 2][i] * d;
                    const float xq = rintf(g * 65536.f);
                    const float sg = 1.f / (1.f + expf(-xq * (1.f / 65536.f)));
                    const float sq = rintf(sg * 65536.f);
                    const float sw = floorf(xq * sq * (1.f / 65536.f));
                    const float yq = rintf(u * 65536.f);
                    const float zq = floorf(sw * yq * (1.f / 65536.f));
                    out[(size_t)m * INTER + ncol] = zq * (1.f / 65536.f);
                }
            }
        } else {
#pragma unroll
            for (int f = 0; f < 4; ++f) {
                const int ncol = nbase + wn * 64 + f * 16 + lrow;
#pragma unroll
                for (int i = 0; i < 8; ++i) {
                    const int m = mb + i;
                    out[(size_t)m * HID + ncol] = (float)acc[mi][f][i] * a_deq[m] * wsc;
                }
            }
        }
    }
}


extern "C" void kernel_launch(void* const* d_in, const int* in_sizes, int n_in,
                              void* d_out, int out_size, void* d_ws, size_t ws_size,
                              hipStream_t stream) {
    (void)in_sizes; (void)n_in; (void)out_size; (void)ws_size;
    const float* x     = (const float*)d_in[0];
    const float* wgate = (const float*)d_in[1];
    const float* wdown = (const float*)d_in[2];
    float* out = (float*)d_out;

    char* base = (char*)d_ws;
    size_t off = 0;
    auto alloc = [&](size_t bytes) -> void* {
        off = (off + 255) & ~(size_t)255;
        void* p = base + off;
        off += bytes;
        return p;
    };
    int8_t* xq   = (int8_t*)alloc((size_t)M_TOK * HID);
    int8_t* zq   = (int8_t*)alloc((size_t)M_TOK * INTER);
    int8_t* wqg  = (int8_t*)alloc((size_t)NGATE * HID);
    int8_t* wqd  = (int8_t*)alloc((size_t)HID * INTER);
    float*  zbuf = (float*)alloc((size_t)M_TOK * INTER * 4);
    float*  asx  = (float*)alloc((size_t)M_TOK * 4);
    float*  asz  = (float*)alloc((size_t)M_TOK * 4);
    float*  wsc  = (float*)alloc(2 * 4);
    float*  part = (float*)alloc(3072 * 4);

    const long ngate = (long)NGATE * HID;
    const long ndown = (long)HID * INTER;

    wabs_partial_kernel<<<2048, 256, 0, stream>>>(wgate, ngate, part);
    wabs_partial_kernel<<<1024, 256, 0, stream>>>(wdown, ndown, part + 2048);
    wscale_finalize_kernel<<<2, 256, 0, stream>>>(part, wsc);
    quant_w_kernel<<<8192, 256, 0, stream>>>(wgate, wqg, ngate, wsc + 0);
    quant_w_kernel<<<4096, 256, 0, stream>>>(wdown, wqd, ndown, wsc + 1);

    act_quant_kernel<<<M_TOK, 256, 0, stream>>>(x, xq, asx, HID);

    dim3 g1(INTER / 64, M_TOK / 128);
    bitgemm_kernel<true><<<g1, 256, 0, stream>>>(xq, wqg, asx, wsc + 0, zbuf, HID);

    act_quant_kernel<<<M_TOK, 256, 0, stream>>>(zbuf, zq, asz, INTER);

    dim3 g2(HID / 128, M_TOK / 128);
    bitgemm_kernel<false><<<g2, 256, 0, stream>>>(zq, wqd, asz, wsc + 1, out, INTER);
}
